// MambaBlock_75797582840220
// MI455X (gfx1250) — matrix operand order repaired and checked by reading, not compiled, not run on hardware
//
#include <hip/hip_runtime.h>

typedef __attribute__((ext_vector_type(16))) _Float16 v16h;
typedef __attribute__((ext_vector_type(8)))  _Float16 v8h;
typedef __attribute__((ext_vector_type(4)))  _Float16 v4h;
typedef __attribute__((ext_vector_type(8)))  float    v8f;

#define D_MODEL 1024
#define TLEN    4096
#define BATCH   4
#define MROWS   (BATCH * TLEN)   // 16384
#define NCH     8
#define CHLEN   (TLEN / NCH)     // 512
#define LSTR    40               // f16 per LDS row: 32 data + 8 pad -> 80B, 16B aligned

__device__ __forceinline__ unsigned lds_off(const void* p) {
    // flat LDS address: low 32 bits are the LDS byte offset (ISA 10.2 aperture rules)
    return (unsigned)(uintptr_t)p;
}

// one 16B chunk, global -> LDS, tracked by ASYNCcnt
__device__ __forceinline__ void async_cp16(unsigned lds, unsigned goff, const _Float16* base) {
    asm volatile("global_load_async_to_lds_b128 %0, %1, %2"
                 :: "v"(lds), "v"(goff), "s"(base)
                 : "memory");
}

#define SHUF16(lo, hi) __builtin_shufflevector(lo, hi, 0,1,2,3,4,5,6,7,8,9,10,11,12,13,14,15)

// C[M,N] = A[M,K]*W[N,K]^T (+bias / +gate epilogue). f16 operands, f32 accum.
// grid (N/128, M/64); 256 threads = 8 wave32 as 2(M) x 4(N); wave tile 32x32.
// GATE: also computes the +D_MODEL column block of W and writes decay|inp in place of u|delta.
template <bool GATE>
__launch_bounds__(256)
__global__ void gemm_wmma(const _Float16* __restrict__ Ah, int lda,
                          const _Float16* __restrict__ Wh, int ldw,
                          const float* __restrict__ bias,
                          const float* __restrict__ Avec,
                          const float* __restrict__ Bp,
                          float* __restrict__ Out, int ldc, int K)
{
    __shared__ __align__(16) _Float16 smA [2][64  * LSTR];
    __shared__ __align__(16) _Float16 smB [2][128 * LSTR];
    __shared__ __align__(16) _Float16 smBd[2][128 * LSTR];   // GATE only

    const int tid   = threadIdx.x;
    const int lane  = tid & 31;
    const int wid   = tid >> 5;
    const int waveM = wid & 1;
    const int waveN = wid >> 1;
    const int mBase = blockIdx.y * 64;
    const int nBase = blockIdx.x * 128;
    const int r     = lane & 15;
    const int hK    = lane >> 4;
    const int nsteps = K / 32;

    // staging chunk coords: A = 64 rows x 4 x 16B, B = 128 rows x 4 x 16B
    const int aRow = tid >> 2, aH = tid & 3;

    auto stage = [&](int ks, int bi) {
        const int k0 = ks * 32;
        {   // A: 1 chunk/thread
            unsigned l = lds_off(&smA[bi][0]) + (unsigned)(aRow * 80 + aH * 16);
            unsigned g = (unsigned)((((mBase + aRow) * lda) + k0 + aH * 8) * 2);
            async_cp16(l, g, Ah);
        }
#pragma unroll
        for (int q = 0; q < 2; ++q) {   // B: 2 chunks/thread
            int cc = tid + q * 256;
            int br = cc >> 2, bh = cc & 3;
            unsigned l = lds_off(&smB[bi][0]) + (unsigned)(br * 80 + bh * 16);
            unsigned g = (unsigned)((((nBase + br) * ldw) + k0 + bh * 8) * 2);
            async_cp16(l, g, Wh);
        }
        if constexpr (GATE) {
#pragma unroll
            for (int q = 0; q < 2; ++q) {   // delta-block of W_in
                int cc = tid + q * 256;
                int br = cc >> 2, bh = cc & 3;
                unsigned l = lds_off(&smBd[bi][0]) + (unsigned)(br * 80 + bh * 16);
                unsigned g = (unsigned)((((D_MODEL + nBase + br) * ldw) + k0 + bh * 8) * 2);
                async_cp16(l, g, Wh);
            }
        }
    };

    v8f accU[2][2], accD[2][2];
#pragma unroll
    for (int i = 0; i < 2; ++i)
#pragma unroll
        for (int j = 0; j < 2; ++j)
#pragma unroll
            for (int e = 0; e < 8; ++e) { accU[i][j][e] = 0.0f; accD[i][j][e] = 0.0f; }

    stage(0, 0);
    if (nsteps > 1) stage(1, 1);

    for (int i = 0; i < nsteps; ++i) {
        const int bi = i & 1;
        // wait for tile i (oldest); leave tile i+1 (C ops/thread) in flight
        if (i + 1 < nsteps) {
            if constexpr (GATE) asm volatile("s_wait_asynccnt 5" ::: "memory");
            else                asm volatile("s_wait_asynccnt 3" ::: "memory");
        } else {
            asm volatile("s_wait_asynccnt 0" ::: "memory");
        }
        __syncthreads();

        // A frags: lane holds row (lane&15); half-lane selects K 16-chunk
        v16h aF[2];
#pragma unroll
        for (int ii = 0; ii < 2; ++ii) {
            const _Float16* p = &smA[bi][(waveM * 32 + ii * 16 + r) * LSTR + hK * 16];
            v8h lo = *(const v8h*)p;
            v8h hi = *(const v8h*)(p + 8);
            aF[ii] = SHUF16(lo, hi);
        }
        // B frags: lane holds column n (lane&15); half-lane selects K 16-chunk
        v16h bU[2], bD[2];
#pragma unroll
        for (int j = 0; j < 2; ++j) {
            const _Float16* p = &smB[bi][(waveN * 32 + j * 16 + r) * LSTR + hK * 16];
            v8h lo = *(const v8h*)p;
            v8h hi = *(const v8h*)(p + 8);
            bU[j] = SHUF16(lo, hi);
            if constexpr (GATE) {
                const _Float16* q = &smBd[bi][(waveN * 32 + j * 16 + r) * LSTR + hK * 16];
                v8h lo2 = *(const v8h*)q;
                v8h hi2 = *(const v8h*)(q + 8);
                bD[j] = SHUF16(lo2, hi2);
            }
        }

#pragma unroll
        for (int ii = 0; ii < 2; ++ii)
#pragma unroll
            for (int j = 0; j < 2; ++j) {
                accU[ii][j] = __builtin_amdgcn_wmma_f32_16x16x32_f16(
                    false, aF[ii], false, bU[j], (short)0, accU[ii][j], false, false);
                if constexpr (GATE)
                    accD[ii][j] = __builtin_amdgcn_wmma_f32_16x16x32_f16(
                        false, aF[ii], false, bD[j], (short)0, accD[ii][j], false, false);
            }

        __syncthreads();
        if (i + 2 < nsteps) stage(i + 2, bi);   // buffer bi is free now
    }

    // epilogue: VGPR e -> row = base + (lane>>4)*8 + e, col = base + (lane&15)
#pragma unroll
    for (int ii = 0; ii < 2; ++ii) {
        const int row = mBase + waveM * 32 + ii * 16 + hK * 8;
#pragma unroll
        for (int j = 0; j < 2; ++j) {
            const int col = nBase + waveN * 32 + j * 16 + r;
            if constexpr (GATE) {
                const float bu = bias[col], bd = bias[D_MODEL + col];
                const float av = Avec[col], bp = Bp[col];
                float* du = Out + (size_t)row * ldc + col;
#pragma unroll
                for (int e = 0; e < 8; ++e) {
                    float u     = accU[ii][j][e] + bu;
                    float dl    = accD[ii][j][e] + bd;
                    float delta = 1.0f / (1.0f + __expf(-dl));
                    du[(size_t)e * ldc]           = __expf(-av * delta);  // decay
                    du[(size_t)e * ldc + D_MODEL] = bp * u;               // inp
                }
            } else {
                const float bb = bias[col];
                float* cp = Out + (size_t)row * ldc + col;
#pragma unroll
                for (int e = 0; e < 8; ++e)
                    cp[(size_t)e * ldc] = accU[ii][j][e] + bb;
            }
        }
    }
}

// f32 -> f16 bulk convert, 4 elements/thread
__global__ void cvt_f16(const float* __restrict__ s, _Float16* __restrict__ d, int n4)
{
    int i = blockIdx.x * 256 + threadIdx.x;
    if (i >= n4) return;
    float4 f = ((const float4*)s)[i];
    v4h h;
    h[0] = (_Float16)f.x; h[1] = (_Float16)f.y; h[2] = (_Float16)f.z; h[3] = (_Float16)f.w;
    ((v4h*)d)[i] = h;
}

// Pass A: per-chunk decay product and local scan end (h0=0)
__global__ void scan_chunks(const float* __restrict__ buf,
                            float* __restrict__ Parr, float* __restrict__ Harr)
{
    const int d = (blockIdx.x & 3) * 256 + threadIdx.x;
    const int c = (blockIdx.x >> 2) & (NCH - 1);
    const int b = blockIdx.x >> 5;
    const float* base = buf + (size_t)(b * TLEN + c * CHLEN) * (2 * D_MODEL);
    float P = 1.0f, h = 0.0f;
    for (int t = 0; t < CHLEN; ++t) {
        float dec = base[(size_t)t * (2 * D_MODEL) + d];
        float in  = base[(size_t)t * (2 * D_MODEL) + D_MODEL + d];
        P *= dec;
        h  = dec * h + in;
    }
    size_t o = (size_t)(b * NCH + c) * D_MODEL + d;
    Parr[o] = P;
    Harr[o] = h;
}

// Pass B: sequential combine; Parr[c] becomes the carry-in h for chunk c
__global__ void scan_carry(float* __restrict__ Parr, const float* __restrict__ Harr)
{
    int idx = blockIdx.x * 256 + threadIdx.x;   // BATCH*D
    int b = idx / D_MODEL, d = idx % D_MODEL;
    float h = 0.0f;
    for (int c = 0; c < NCH; ++c) {
        size_t o = (size_t)(b * NCH + c) * D_MODEL + d;
        float P  = Parr[o];
        float he = Harr[o];
        Parr[o] = h;
        h = P * h + he;
    }
}

// Pass C: apply carry, emit y directly as f16 (GEMM2 operand)
__global__ void scan_apply(const float* __restrict__ buf, const float* __restrict__ Carry,
                           _Float16* __restrict__ yh)
{
    const int d = (blockIdx.x & 3) * 256 + threadIdx.x;
    const int c = (blockIdx.x >> 2) & (NCH - 1);
    const int b = blockIdx.x >> 5;
    const float*    base = buf + (size_t)(b * TLEN + c * CHLEN) * (2 * D_MODEL);
    _Float16*       yb   = yh  + (size_t)(b * TLEN + c * CHLEN) * D_MODEL;
    float h = Carry[(size_t)(b * NCH + c) * D_MODEL + d];
    for (int t = 0; t < CHLEN; ++t) {
        float dec = base[(size_t)t * (2 * D_MODEL) + d];
        float in  = base[(size_t)t * (2 * D_MODEL) + D_MODEL + d];
        h = dec * h + in;
        yb[(size_t)t * D_MODEL + d] = (_Float16)h;
    }
}

extern "C" void kernel_launch(void* const* d_in, const int* in_sizes, int n_in,
                              void* d_out, int out_size, void* d_ws, size_t ws_size,
                              hipStream_t stream)
{
    (void)in_sizes; (void)n_in; (void)out_size; (void)ws_size;
    const float* x     = (const float*)d_in[0];
    const float* W_in  = (const float*)d_in[1];
    const float* b_in  = (const float*)d_in[2];
    const float* Avec  = (const float*)d_in[3];
    const float* Bp    = (const float*)d_in[4];
    const float* W_out = (const float*)d_in[5];
    const float* b_out = (const float*)d_in[6];
    float* out = (float*)d_out;

    char* w = (char*)d_ws;
    float*    buf = (float*)w;     w += (size_t)MROWS * 2 * D_MODEL * 4;   // decay|inp
    _Float16* xh  = (_Float16*)w;  w += (size_t)MROWS * D_MODEL * 2;
    _Float16* Wih = (_Float16*)w;  w += (size_t)2 * D_MODEL * D_MODEL * 2;
    _Float16* Woh = (_Float16*)w;  w += (size_t)D_MODEL * D_MODEL * 2;
    _Float16* yh  = (_Float16*)w;  w += (size_t)MROWS * D_MODEL * 2;
    float*    Parr = (float*)w;    w += (size_t)BATCH * NCH * D_MODEL * 4;
    float*    Harr = (float*)w;

    dim3 blk(256);

    // precision downcast of GEMM operands
    cvt_f16<<<(MROWS * D_MODEL / 4) / 256, blk, 0, stream>>>(x, xh, MROWS * D_MODEL / 4);
    cvt_f16<<<(2 * D_MODEL * D_MODEL / 4) / 256, blk, 0, stream>>>(W_in, Wih, 2 * D_MODEL * D_MODEL / 4);
    cvt_f16<<<(D_MODEL * D_MODEL / 4) / 256, blk, 0, stream>>>(W_out, Woh, D_MODEL * D_MODEL / 4);

    // GEMM1 + fused gate -> buf = decay|inp
    gemm_wmma<true><<<dim3(D_MODEL / 128, MROWS / 64), blk, 0, stream>>>(
        xh, D_MODEL, Wih, D_MODEL, b_in, Avec, Bp, buf, 2 * D_MODEL, D_MODEL);

    // chunked first-order scan
    scan_chunks<<<BATCH * NCH * (D_MODEL / 256), blk, 0, stream>>>(buf, Parr, Harr);
    scan_carry<<<(BATCH * D_MODEL) / 256, blk, 0, stream>>>(Parr, Harr);
    scan_apply<<<BATCH * NCH * (D_MODEL / 256), blk, 0, stream>>>(buf, Parr, yh);

    // GEMM2 -> out
    gemm_wmma<false><<<dim3(D_MODEL / 128, MROWS / 64), blk, 0, stream>>>(
        yh, D_MODEL, Woh, D_MODEL, b_out, nullptr, nullptr, out, D_MODEL, D_MODEL);
}
